// MaxViT_17678085390570
// MI455X (gfx1250) — compile-verified, matrix operand order checked by reading, not run on hardware
//
#include <hip/hip_runtime.h>
#include <math.h>

// ---------------------------------------------------------------------------
// Types / WMMA helpers (CDNA5 wave32, V_WMMA_F32_16X16X32_F16)
// ---------------------------------------------------------------------------
typedef __attribute__((ext_vector_type(16))) _Float16 v16h;
typedef __attribute__((ext_vector_type(8)))  float    v8f;

#define WMMA16(a, b, c) \
  __builtin_amdgcn_wmma_f32_16x16x32_f16(false, (a), false, (b), (short)0, (c), false, false)

static constexpr int ACT_NONE = 0, ACT_GELU = 1, ACT_SILU = 2, ACT_SIGMOID = 3;

__device__ __forceinline__ float gelu_f(float x) {
  return 0.5f * x * (1.0f + erff(x * 0.7071067811865475f));
}
__device__ __forceinline__ float sigmoid_f(float x) {
  return 1.0f / (1.0f + __expf(-x));
}

// K index held by half `h` (0..15) of a lane with hi = lane>>4, per ISA A/B
// 16-bit fragment layout: V0..V3 cover K = hi*8 + 0..7, V4..V7 cover +16.
__device__ __forceinline__ int frag_k(int h, int hi) {
  return ((h >> 3) << 4) + (hi << 3) + (h & 7);
}

// Row-major [row][32+pad] fragment load: lane's 16 halves are 2 contiguous
// 8-half runs -> lowers to two ds_load_b128.
__device__ __forceinline__ v16h load_row_h16(const _Float16* T, int ld, int r0, int k0, int lane) {
  int r = lane & 15, hi = lane >> 4;
  v16h a;
#pragma unroll
  for (int h = 0; h < 16; ++h) a[h] = T[(r0 + r) * ld + k0 + frag_k(h, hi)];
  return a;
}

// ---------------------------------------------------------------------------
// LDS-tiled, double-buffered WMMA GEMM: C[M,N] = epi(A[M,K] @ B[K,N]).
// 256 threads (8 waves) -> 64x128 tile; wave = 32x32 sub-tile (4 wmma/K-step).
// A staged [m][k] (64x40), B staged transposed [n][k] (128x40) so both
// fragment reads are contiguous ds_load_b128. Requires K % 32 == 0.
// Bounds via clamp+select at copy granularity; batched loads -> single wait.
// Epilogue: y = acc*scaleM[m] + biasM[m] + biasN[n]; act; y += res; (all opt.)
// ---------------------------------------------------------------------------
__global__ void __launch_bounds__(256)
gemm_wmma_kernel(const float* __restrict__ A, const float* __restrict__ B,
                 float* __restrict__ C, const float* __restrict__ res,
                 const float* __restrict__ scaleM, const float* __restrict__ biasM,
                 const float* __restrict__ biasN,
                 int M, int N, int K, int lda, int ldb, int ldc,
                 long sAe, long sBe, long sCe, long sRe, int act) {
  constexpr int BM = 64, BN = 128, LDT = 40;
  __shared__ _Float16 shA[2][BM * LDT];    // [m][k]
  __shared__ _Float16 shBt[2][BN * LDT];   // [n][k] (transposed)

  int tid = threadIdx.x, lane = tid & 31, wave = tid >> 5;
  int nt = (N + BN - 1) / BN;
  int mi = blockIdx.x / nt, ni = blockIdx.x % nt;
  int m0 = mi * BM, n0 = ni * BN;
  long zb = blockIdx.z;
  const float* Ab = A + zb * sAe;
  const float* Bb = B + zb * sBe;
  float* Cb = C + zb * sCe;
  const float* Rb = res ? res + zb * sRe : (const float*)0;

  int wm = wave >> 2;   // 0..1 : wave row    (32 rows)
  int wn = wave & 3;    // 0..3 : wave column (32 cols)

  v8f acc00 = {}, acc01 = {}, acc10 = {}, acc11 = {};

  float ra[8];    // A: 64x32 / 256 thr
  float rb[16];   // B: 32x128 / 256 thr

  // ---- staging helpers (batched independent loads, then convert+store) ----
  auto load_regs = [&](int k0) {
#pragma unroll
    for (int j = 0; j < 8; ++j) {
      int i = tid + j * 256;
      int r = i >> 5, c = i & 31;
      int m = m0 + r;
      long mc = (m < M) ? m : (M - 1);
      ra[j] = Ab[mc * (long)lda + k0 + c];
    }
#pragma unroll
    for (int j = 0; j < 16; ++j) {
      int i = tid + j * 256;
      int r = i >> 7, c = i & 127;
      int n = n0 + c;
      long nc = (n < N) ? n : (N - 1);
      rb[j] = Bb[(long)(k0 + r) * ldb + nc];
    }
  };
  auto store_lds = [&](int p) {
#pragma unroll
    for (int j = 0; j < 8; ++j) {
      int i = tid + j * 256;
      int r = i >> 5, c = i & 31;
      int m = m0 + r;
      shA[p][r * LDT + c] = (m < M) ? (_Float16)ra[j] : (_Float16)0.f;
    }
#pragma unroll
    for (int j = 0; j < 16; ++j) {
      int i = tid + j * 256;
      int r = i >> 7, c = i & 127;
      int n = n0 + c;
      shBt[p][c * LDT + r] = (n < N) ? (_Float16)rb[j] : (_Float16)0.f;
    }
  };
  auto compute = [&](int p) {
    v16h a0 = load_row_h16(shA[p], LDT, wm * 32, 0, lane);
    v16h a1 = load_row_h16(shA[p], LDT, wm * 32 + 16, 0, lane);
    v16h b0 = load_row_h16(shBt[p], LDT, wn * 32, 0, lane);
    v16h b1 = load_row_h16(shBt[p], LDT, wn * 32 + 16, 0, lane);
    acc00 = WMMA16(a0, b0, acc00);
    acc01 = WMMA16(a0, b1, acc01);
    acc10 = WMMA16(a1, b0, acc10);
    acc11 = WMMA16(a1, b1, acc11);
  };

  // ---- software pipeline: 1 barrier per K-step ----
  load_regs(0);
  store_lds(0);
  int p = 0;
  for (int k0 = 32; k0 < K; k0 += 32) {
    load_regs(k0);      // next tile's global loads in flight during compute
    __syncthreads();    // buf[p] visible to all
    compute(p);
    store_lds(1 - p);   // no thread reads buf[1-p] this iteration
    p ^= 1;
  }
  __syncthreads();
  compute(p);

  // ---- epilogue ----
  int c0 = lane & 15, hi = lane >> 4;
#pragma unroll
  for (int tm = 0; tm < 2; ++tm) {
#pragma unroll
    for (int tn = 0; tn < 2; ++tn) {
      v8f acc = tm ? (tn ? acc11 : acc10) : (tn ? acc01 : acc00);
#pragma unroll
      for (int r = 0; r < 8; ++r) {
        int m = m0 + wm * 32 + tm * 16 + r + hi * 8;
        int n = n0 + wn * 32 + tn * 16 + c0;
        if (m < M && n < N) {
          float y = acc[r];
          if (scaleM) y *= scaleM[m];
          if (biasM)  y += biasM[m];
          if (biasN)  y += biasN[n];
          if (act == ACT_GELU)         y = gelu_f(y);
          else if (act == ACT_SILU)    y *= sigmoid_f(y);
          else if (act == ACT_SIGMOID) y = sigmoid_f(y);
          if (Rb) y += Rb[(long)m * ldc + n];
          Cb[(long)m * ldc + n] = y;
        }
      }
    }
  }
}

// ---------------------------------------------------------------------------
// Fused window attention: one workgroup (128 thr = 4 waves) per (window,head).
// qkv: [tokens, 3D]; out: [tokens, D]; rel: [(2W-1)^2, heads]. n = 49 -> pad 64.
// All LDS tiles laid out so fragment reads are contiguous (ds_load_b128):
//   sq, sk: [token][feat] (64x40), svvT: [feat][token] (32x72),
//   sp: [row][col] (64x72), ssim padded to 65 f32/row (conflict-free softmax).
// ---------------------------------------------------------------------------
__global__ void __launch_bounds__(128)
attn_kernel(const float* __restrict__ qkv, float* __restrict__ out,
            const float* __restrict__ rel, int heads, int D) {
  int win = blockIdx.x / heads;
  int head = blockIdx.x % heads;
  long tok0 = (long)win * 49;
  int tid = threadIdx.x, lane = tid & 31, wave = tid >> 5;
  int D3 = 3 * D;

  __shared__ _Float16 sq[64 * 40];
  __shared__ _Float16 sk[64 * 40];
  __shared__ _Float16 svvT[32 * 72];
  __shared__ float    ssim[64 * 65];
  __shared__ _Float16 sp[64 * 72];

  // Batched q/k/v loads (independent, single wait), then convert+store.
  float vq[16], vk[16], vv[16];
#pragma unroll
  for (int j = 0; j < 16; ++j) {
    int idx = tid + j * 128;
    int r = idx >> 5, c = idx & 31;
    long rr = (r < 49) ? r : 48;
    const float* base = qkv + (tok0 + rr) * D3 + head * 32 + c;
    vq[j] = base[0];
    vk[j] = base[D];
    vv[j] = base[2 * D];
  }
#pragma unroll
  for (int j = 0; j < 16; ++j) {
    int idx = tid + j * 128;
    int r = idx >> 5, c = idx & 31;
    bool ok = (r < 49);
    sq[r * 40 + c]   = ok ? (_Float16)(vq[j] * 0.17677669529663687f) : (_Float16)0.f;
    sk[r * 40 + c]   = ok ? (_Float16)vk[j] : (_Float16)0.f;
    svvT[c * 72 + r] = ok ? (_Float16)vv[j] : (_Float16)0.f;
  }
  __syncthreads();

  // sim = q @ k^T  (64x64, 4x4 tiles of 16x16, K=32). B[k][n] = sk[n][k].
  for (int t = wave; t < 16; t += (blockDim.x >> 5)) {
    int mi = t >> 2, ni = t & 3;
    v16h a = load_row_h16(sq, 40, mi * 16, 0, lane);
    v16h b = load_row_h16(sk, 40, ni * 16, 0, lane);
    v8f acc = {};
    acc = WMMA16(a, b, acc);
    int c0 = lane & 15, hi = lane >> 4;
#pragma unroll
    for (int r = 0; r < 8; ++r)
      ssim[(mi * 16 + r + hi * 8) * 65 + ni * 16 + c0] = acc[r];
  }
  __syncthreads();

  // + rel-pos bias, row softmax -> sp (f16 probs, zero padded)
  for (int i = tid; i < 64; i += blockDim.x) {
    if (i < 49) {
      int r1 = i / 7, c1 = i % 7;
      float vals[49];
      float mx = -1e30f;
      for (int j = 0; j < 49; ++j) {
        int r2 = j / 7, c2 = j % 7;
        int ridx = (r1 - r2 + 6) * 13 + (c1 - c2 + 6);
        float s = ssim[i * 65 + j] + rel[ridx * heads + head];
        vals[j] = s;
        mx = fmaxf(mx, s);
      }
      float sum = 0.f;
      for (int j = 0; j < 49; ++j) { float e = __expf(vals[j] - mx); vals[j] = e; sum += e; }
      float inv = 1.0f / sum;
      for (int j = 0; j < 49; ++j) sp[i * 72 + j] = (_Float16)(vals[j] * inv);
      for (int j = 49; j < 64; ++j) sp[i * 72 + j] = (_Float16)0.f;
    } else {
      for (int j = 0; j < 64; ++j) sp[i * 72 + j] = (_Float16)0.f;
    }
  }
  __syncthreads();

  // out = sp(64x64) @ v(64x32); 4x2 tiles, K loop of 2. B[k][n] = svvT[n][k].
  for (int t = wave; t < 8; t += (blockDim.x >> 5)) {
    int mi = t >> 1, ni = t & 1;
    v8f acc = {};
    for (int k0 = 0; k0 < 64; k0 += 32) {
      v16h a = load_row_h16(sp, 72, mi * 16, k0, lane);
      v16h b = load_row_h16(svvT, 72, ni * 16, k0, lane);
      acc = WMMA16(a, b, acc);
    }
    int c0 = lane & 15, hi = lane >> 4;
#pragma unroll
    for (int r = 0; r < 8; ++r) {
      int m = mi * 16 + r + hi * 8;
      int n = ni * 16 + c0;
      if (m < 49) out[(tok0 + m) * D + head * 32 + n] = acc[r];
    }
  }
}

// ---------------------------------------------------------------------------
// Elementwise / reduction helpers
// ---------------------------------------------------------------------------
__global__ void bnfold_kernel(const float* g, const float* b, const float* m,
                              const float* v, float* scale, float* bias, int C) {
  int i = blockIdx.x * blockDim.x + threadIdx.x;
  if (i < C) {
    float inv = g[i] * rsqrtf(v[i] + 1e-5f);
    scale[i] = inv;
    bias[i] = b[i] - m[i] * inv;
  }
}

__global__ void dwconv_kernel(const float* __restrict__ x, const float* __restrict__ w,
                              float* __restrict__ y, const float* __restrict__ scale,
                              const float* __restrict__ bias,
                              int C, int H, int W, int Ho, int Wo, int stride, long tot) {
  long i = (long)blockIdx.x * blockDim.x + threadIdx.x;
  if (i >= tot) return;
  int wo = (int)(i % Wo); long r = i / Wo;
  int ho = (int)(r % Ho); r /= Ho;
  int c = (int)(r % C);   int n = (int)(r / C);
  const float* xb = x + ((long)n * C + c) * H * W;
  const float* wc = w + c * 9;
  float acc = 0.f;
#pragma unroll
  for (int kh = 0; kh < 3; ++kh) {
    int h = ho * stride - 1 + kh;
    if ((unsigned)h >= (unsigned)H) continue;
#pragma unroll
    for (int kw = 0; kw < 3; ++kw) {
      int ww = wo * stride - 1 + kw;
      if ((unsigned)ww >= (unsigned)W) continue;
      acc += xb[h * W + ww] * wc[kh * 3 + kw];
    }
  }
  y[i] = gelu_f(acc * scale[c] + bias[c]);
}

__global__ void conv3x3_kernel(const float* __restrict__ x, const float* __restrict__ w,
                               const float* __restrict__ bias, float* __restrict__ y,
                               int Ci, int Co, int H, int W, int Ho, int Wo,
                               int stride, long tot) {
  long i = (long)blockIdx.x * blockDim.x + threadIdx.x;
  if (i >= tot) return;
  int wo = (int)(i % Wo); long r = i / Wo;
  int ho = (int)(r % Ho); r /= Ho;
  int co = (int)(r % Co); int n = (int)(r / Co);
  float acc = bias[co];
  for (int ci = 0; ci < Ci; ++ci) {
    const float* xb = x + ((long)n * Ci + ci) * H * W;
    const float* wb = w + ((long)co * Ci + ci) * 9;
#pragma unroll
    for (int kh = 0; kh < 3; ++kh) {
      int h = ho * stride - 1 + kh;
      if ((unsigned)h >= (unsigned)H) continue;
#pragma unroll
      for (int kw = 0; kw < 3; ++kw) {
        int ww = wo * stride - 1 + kw;
        if ((unsigned)ww >= (unsigned)W) continue;
        acc += xb[h * W + ww] * wb[kh * 3 + kw];
      }
    }
  }
  y[i] = acc;
}

// im2col for 3x3 s1 p1: col[(ci*9+kh*3+kw)*HW + oh*W+ow]
__global__ void im2col3x3_kernel(const float* __restrict__ x, float* __restrict__ col,
                                 int C, int H, int W, long tot) {
  long i = (long)blockIdx.x * blockDim.x + threadIdx.x;
  if (i >= tot) return;
  int hw = H * W;
  int cix = (int)(i % hw); long r = i / hw;
  int kk = (int)(r % 9);  int ci = (int)(r / 9);
  int kh = kk / 3, kw = kk % 3;
  int oh = cix / W, ow = cix % W;
  int ih = oh - 1 + kh, iw = ow - 1 + kw;
  float v = ((unsigned)ih < (unsigned)H && (unsigned)iw < (unsigned)W)
            ? x[((long)ci * H + ih) * W + iw] : 0.f;
  col[i] = v;
}

__global__ void pool_kernel(const float* __restrict__ x, float* __restrict__ s, int HW) {
  int nc = blockIdx.x;
  const float* xb = x + (long)nc * HW;
  __shared__ float red[256];
  float acc = 0.f;
  for (int i = threadIdx.x; i < HW; i += blockDim.x) acc += xb[i];
  red[threadIdx.x] = acc;
  __syncthreads();
  for (int o = 128; o > 0; o >>= 1) {
    if (threadIdx.x < (unsigned)o) red[threadIdx.x] += red[threadIdx.x + o];
    __syncthreads();
  }
  if (threadIdx.x == 0) s[nc] = red[0] / (float)HW;
}

__global__ void semul_kernel(float* __restrict__ x, const float* __restrict__ s,
                             long HW, long tot) {
  long i = (long)blockIdx.x * blockDim.x + threadIdx.x;
  if (i >= tot) return;
  x[i] *= s[i / HW];
}

__global__ void ln_kernel(const float* __restrict__ x, float* __restrict__ y,
                          const float* __restrict__ g, const float* __restrict__ b,
                          int T, int D) {
  int t = blockIdx.x;
  if (t >= T) return;
  const float* xr = x + (long)t * D;
  float* yr = y + (long)t * D;
  __shared__ float red[64];
  int tid = threadIdx.x;
  float s = 0.f;
  for (int i = tid; i < D; i += blockDim.x) s += xr[i];
  red[tid] = s;
  __syncthreads();
  for (int o = 32; o > 0; o >>= 1) {
    if (tid < o) red[tid] += red[tid + o];
    __syncthreads();
  }
  float mu = red[0] / (float)D;
  __syncthreads();
  float v = 0.f;
  for (int i = tid; i < D; i += blockDim.x) { float d = xr[i] - mu; v += d * d; }
  red[tid] = v;
  __syncthreads();
  for (int o = 32; o > 0; o >>= 1) {
    if (tid < o) red[tid] += red[tid + o];
    __syncthreads();
  }
  float inv = rsqrtf(red[0] / (float)D + 1e-5f);
  for (int i = tid; i < D; i += blockDim.x) yr[i] = (xr[i] - mu) * inv * g[i] + b[i];
}

// Window (mode 0: block-local) / grid (mode 1: dilated) token permutes.
__device__ __forceinline__ long win_src_index(long ti, int d, int D, int H, int W,
                                              int X, int Y, int mode) {
  int within = (int)(ti % 49); long wb = ti / 49;
  int yy = (int)(wb % Y); wb /= Y;
  int xx = (int)(wb % X); int b = (int)(wb / X);
  int w1 = within / 7, w2 = within % 7;
  int h, w;
  if (mode == 0) { h = xx * 7 + w1; w = yy * 7 + w2; }
  else           { h = w1 * X + xx; w = w2 * Y + yy; }
  return (((long)b * D + d) * H + h) * W + w;
}
__global__ void gather_kernel(const float* __restrict__ x, float* __restrict__ t,
                              int D, int H, int W, int X, int Y, int mode, long tot) {
  long i = (long)blockIdx.x * blockDim.x + threadIdx.x;
  if (i >= tot) return;
  int d = (int)(i % D);
  long ti = i / D;
  t[i] = x[win_src_index(ti, d, D, H, W, X, Y, mode)];
}
__global__ void scatter_kernel(float* __restrict__ x, const float* __restrict__ t,
                               int D, int H, int W, int X, int Y, int mode, long tot) {
  long i = (long)blockIdx.x * blockDim.x + threadIdx.x;
  if (i >= tot) return;
  int d = (int)(i % D);
  long ti = i / D;
  x[win_src_index(ti, d, D, H, W, X, Y, mode)] = t[i];
}

// ---------------------------------------------------------------------------
// Host orchestration
// ---------------------------------------------------------------------------
// Pytree leaf offsets within each block (JAX sorted-dict-key flatten order):
enum {
  A_REL1 = 0, A_WOUT1 = 1, A_WQKV1 = 2,
  A_REL2 = 3, A_WOUT2 = 4, A_WQKV2 = 5,
  FF1_B1 = 6, FF1_B2 = 7, FF1_W1 = 8, FF1_W2 = 9,
  FF2_B1 = 10, FF2_B2 = 11, FF2_W1 = 12, FF2_W2 = 13,
  LNA1_B = 14, LNA1_G = 15, LNA2_B = 16, LNA2_G = 17,
  LNF1_B = 18, LNF1_G = 19, LNF2_B = 20, LNF2_G = 21,
  BN1_B = 22, BN1_G = 23, BN1_M = 24, BN1_V = 25,
  BN2_B = 26, BN2_G = 27, BN2_M = 28, BN2_V = 29,
  BN3_B = 30, BN3_G = 31, BN3_M = 32, BN3_V = 33,
  MB_C1 = 34, MB_C2 = 35, MB_DW = 36, MB_SE1 = 37, MB_SE2 = 38,
  BLK_LEAVES = 39
};

extern "C" void kernel_launch(void* const* d_in, const int* in_sizes, int n_in,
                              void* d_out, int out_size, void* d_ws, size_t ws_size,
                              hipStream_t stream) {
  (void)out_size; (void)ws_size;

  // Locate the image input by its unique flat size; params are the remaining
  // 320 leaves in sorted-pytree order (blocks[0..7], head, stem).
  int ix = 0;
  for (int i = 0; i < n_in; ++i) {
    if (in_sizes[i] == 4 * 3 * 224 * 224) { ix = i; break; }
  }
  const float* x_in = (const float*)d_in[ix];
  const int pbase = (ix == 0) ? 1 : 0;
  auto BL = [&](int b, int off) -> const float* {
    return (const float*)d_in[pbase + b * BLK_LEAVES + off];
  };
  const int hbase = pbase + 8 * BLK_LEAVES;
  const float* head_b    = (const float*)d_in[hbase + 0];
  const float* head_bias = (const float*)d_in[hbase + 1];
  const float* head_g    = (const float*)d_in[hbase + 2];
  const float* head_w    = (const float*)d_in[hbase + 3];
  const float* stem_b1 = (const float*)d_in[hbase + 4];
  const float* stem_b2 = (const float*)d_in[hbase + 5];
  const float* stem_c1 = (const float*)d_in[hbase + 6];
  const float* stem_c2 = (const float*)d_in[hbase + 7];

  // -------- workspace carve (f32 elems, 256B aligned) --------
  size_t off = 0;
  auto alloc = [&](size_t elems) -> float* {
    float* p = (float*)((char*)d_ws + off * sizeof(float));
    off += (elems + 63) & ~(size_t)63;
    return p;
  };
  float* bufX  = alloc(4L * 64 * 112 * 112);   // ping NCHW
  float* bufY  = alloc(4L * 64 * 112 * 112);   // pong NCHW
  float* bufH  = alloc(4L * 256 * 112 * 112);  // mbconv hidden / im2col scratch
  float* bufH2 = alloc(4L * 256 * 56 * 56);    // dw out
  float* T0    = alloc(802816);                // tokens x D
  float* T1    = alloc(802816);
  float* QKV   = alloc(802816L * 3);
  float* Fb    = alloc(802816L * 4);           // FFN hidden
  float* Ob    = alloc(802816);                // attn out
  float* SEs   = alloc(4 * 2048);
  float* SEs1  = alloc(4 * 512);
  float* SEs2  = alloc(4 * 2048);
  float* fS1 = alloc(2048); float* fB1 = alloc(2048);
  float* fS2 = alloc(2048); float* fB2 = alloc(2048);
  float* pooled = alloc(4 * 512);
  float* pooledln = alloc(4 * 512);

  auto gemm = [&](const float* A, const float* B, float* C, const float* res,
                  const float* sM, const float* bM, const float* bN,
                  int M, int N, int K, int lda, int ldb, int ldc,
                  long sAe, long sBe, long sCe, long sRe, int act, int batches) {
    int mt = (M + 63) / 64, nt = (N + 127) / 128;
    dim3 g((unsigned)(mt * nt), 1, (unsigned)batches);
    gemm_wmma_kernel<<<g, 256, 0, stream>>>(A, B, C, res, sM, bM, bN,
                                            M, N, K, lda, ldb, ldc,
                                            sAe, sBe, sCe, sRe, act);
  };
  auto grid1 = [](long tot) { return dim3((unsigned)((tot + 255) / 256)); };

  // -------- stem --------
  {
    long tot1 = 4L * 64 * 112 * 112;
    // conv1: 3x3 s2 p1 (3->64), direct (K=27, cheap)
    conv3x3_kernel<<<grid1(tot1), 256, 0, stream>>>(x_in, stem_c1, stem_b1, bufY,
                                                    3, 64, 224, 224, 112, 112, 2, tot1);
    // conv2: 3x3 s1 p1 (64->64) as im2col (K = 576 = 18*32) + WMMA GEMM
    long coltot = 576L * 112 * 112;
    for (int n = 0; n < 4; ++n) {
      const float* src = bufY + (long)n * 64 * 112 * 112;
      float* dst = bufX + (long)n * 64 * 112 * 112;
      im2col3x3_kernel<<<grid1(coltot), 256, 0, stream>>>(src, bufH, 64, 112, 112, coltot);
      gemm(stem_c2, bufH, dst, nullptr, nullptr, stem_b2, nullptr,
           64, 112 * 112, 576, 576, 112 * 112, 112 * 112, 0, 0, 0, 0, ACT_NONE, 1);
    }
  }

  // -------- attention + FFN phase (shared by block & grid attention) --------
  auto attn_phase = [&](float* X, int D, int H, int mode,
                        const float* lag, const float* lab,
                        const float* rel, const float* wqkv, const float* wout,
                        const float* lfg, const float* lfb,
                        const float* w1, const float* b1,
                        const float* w2, const float* b2) {
    int Xt = H / 7, Yt = H / 7;
    long tokens = 4L * Xt * Yt * 49;
    int heads = D / 32;
    long tot = tokens * D;
    gather_kernel<<<grid1(tot), 256, 0, stream>>>(X, T0, D, H, H, Xt, Yt, mode, tot);
    ln_kernel<<<dim3((unsigned)tokens), 64, 0, stream>>>(T0, T1, lag, lab, (int)tokens, D);
    gemm(T1, wqkv, QKV, nullptr, nullptr, nullptr, nullptr,
         (int)tokens, 3 * D, D, D, 3 * D, 3 * D, 0, 0, 0, 0, ACT_NONE, 1);
    attn_kernel<<<dim3((unsigned)(tokens / 49 * heads)), 128, 0, stream>>>(QKV, Ob, rel, heads, D);
    gemm(Ob, wout, T0, T0, nullptr, nullptr, nullptr,
         (int)tokens, D, D, D, D, D, 0, 0, 0, 0, ACT_NONE, 1);
    ln_kernel<<<dim3((unsigned)tokens), 64, 0, stream>>>(T0, T1, lfg, lfb, (int)tokens, D);
    gemm(T1, w1, Fb, nullptr, nullptr, nullptr, b1,
         (int)tokens, 4 * D, D, D, 4 * D, 4 * D, 0, 0, 0, 0, ACT_GELU, 1);
    gemm(Fb, w2, T0, T0, nullptr, nullptr, b2,
         (int)tokens, D, 4 * D, 4 * D, D, D, 0, 0, 0, 0, ACT_NONE, 1);
    scatter_kernel<<<grid1(tot), 256, 0, stream>>>(X, T0, D, H, H, Xt, Yt, mode, tot);
  };

  // -------- blocks --------
  const int cfg_din[8]  = { 64, 64,  64, 128, 128, 256, 256, 512 };
  const int cfg_dout[8] = { 64, 64, 128, 128, 256, 256, 512, 512 };
  const int cfg_down[8] = {  1,  0,   1,   0,   1,   0,   1,   0 };
  int Hin = 112;

  for (int b = 0; b < 8; ++b) {
    int din = cfg_din[b], dout = cfg_dout[b], down = cfg_down[b];
    int residual = (din == dout) && !down;
    int hid = 4 * dout;
    int H = Hin, Ho = down ? H / 2 : H;
    int HW = H * H, HWo = Ho * Ho;

    // c1 (1x1 conv, BN1 folded, GELU): per-batch GEMM hid x HW
    bnfold_kernel<<<dim3((hid + 255) / 256), 256, 0, stream>>>(
        BL(b, BN1_G), BL(b, BN1_B), BL(b, BN1_M), BL(b, BN1_V), fS1, fB1, hid);
    gemm(BL(b, MB_C1), bufX, bufH, nullptr, fS1, fB1, nullptr,
         hid, HW, din, din, HW, HW, 0, (long)din * HW, (long)hid * HW, 0, ACT_GELU, 4);

    // dw 3x3 (BN2 folded, GELU)
    bnfold_kernel<<<dim3((hid + 255) / 256), 256, 0, stream>>>(
        BL(b, BN2_G), BL(b, BN2_B), BL(b, BN2_M), BL(b, BN2_V), fS2, fB2, hid);
    long totdw = 4L * hid * HWo;
    dwconv_kernel<<<grid1(totdw), 256, 0, stream>>>(bufH, BL(b, MB_DW), bufH2, fS2, fB2,
                                                    hid, H, H, Ho, Ho, down ? 2 : 1, totdw);

    // squeeze-excitation
    pool_kernel<<<dim3(4 * hid), 256, 0, stream>>>(bufH2, SEs, HWo);
    gemm(SEs, BL(b, MB_SE1), SEs1, nullptr, nullptr, nullptr, nullptr,
         4, hid / 4, hid, hid, hid / 4, hid / 4, 0, 0, 0, 0, ACT_SILU, 1);
    gemm(SEs1, BL(b, MB_SE2), SEs2, nullptr, nullptr, nullptr, nullptr,
         4, hid, hid / 4, hid / 4, hid, hid, 0, 0, 0, 0, ACT_SIGMOID, 1);
    semul_kernel<<<grid1(totdw), 256, 0, stream>>>(bufH2, SEs2, HWo, totdw);

    // c2 (1x1 conv, BN3 folded, optional residual)
    bnfold_kernel<<<dim3((dout + 255) / 256), 256, 0, stream>>>(
        BL(b, BN3_G), BL(b, BN3_B), BL(b, BN3_M), BL(b, BN3_V), fS1, fB1, dout);
    const float* res = residual ? bufX : (const float*)0;
    gemm(BL(b, MB_C2), bufH2, bufY, res, fS1, fB1, nullptr,
         dout, HWo, hid, hid, HWo, HWo,
         0, (long)hid * HWo, (long)dout * HWo, (long)dout * HWo, ACT_NONE, 4);
    { float* t = bufX; bufX = bufY; bufY = t; }

    // block (window) attention + FFN, then grid attention + FFN
    attn_phase(bufX, dout, Ho, 0,
               BL(b, LNA1_G), BL(b, LNA1_B),
               BL(b, A_REL1), BL(b, A_WQKV1), BL(b, A_WOUT1),
               BL(b, LNF1_G), BL(b, LNF1_B),
               BL(b, FF1_W1), BL(b, FF1_B1), BL(b, FF1_W2), BL(b, FF1_B2));
    attn_phase(bufX, dout, Ho, 1,
               BL(b, LNA2_G), BL(b, LNA2_B),
               BL(b, A_REL2), BL(b, A_WQKV2), BL(b, A_WOUT2),
               BL(b, LNF2_G), BL(b, LNF2_B),
               BL(b, FF2_W1), BL(b, FF2_B1), BL(b, FF2_W2), BL(b, FF2_B2));
    Hin = Ho;
  }

  // -------- head: global pool -> LN -> linear --------
  pool_kernel<<<dim3(4 * 512), 256, 0, stream>>>(bufX, pooled, 7 * 7);
  ln_kernel<<<dim3(4), 64, 0, stream>>>(pooled, pooledln, head_g, head_b, 4, 512);
  gemm(pooledln, head_w, (float*)d_out, nullptr, nullptr, nullptr, head_bias,
       4, 1000, 512, 512, 1000, 1000, 0, 0, 0, 0, ACT_NONE, 1);
}
